// DeformableTransformer_60516089200682
// MI455X (gfx1250) — matrix operand order repaired and checked by reading, not compiled, not run on hardware
//
#include <hip/hip_runtime.h>
#include <hip/hip_bf16.h>
#include <stdint.h>

// ---------------------------------------------------------------------------
// Deformable-DETR decoder for MI455X (gfx1250, wave32, WMMA).
// Dense algebra: v_wmma_f32_16x16x32_f16 (f16 in, f32 acc).
// GEMM staging: global_load_async_to_lds_b128 double-buffered (ASYNCcnt).
// ---------------------------------------------------------------------------

#define D      256
#define HEADS  8
#define DH     32
#define LEVELS 4
#define POINTS 4
#define NLAY   6
#define DFF    1024
#define BSZ    4
#define NQ     900
#define SEQLEN 19560            // 92*160 + 46*80 + 23*40 + 12*20
#define NTOK   (BSZ * NQ)       // 3600
#define NSRC   (BSZ * SEQLEN)   // 78240
#define EPSV   1e-5f
#define ATT_SCALE 0.1767766952966369f  // 1/sqrt(32)

typedef __attribute__((ext_vector_type(16))) _Float16 v16h;
typedef __attribute__((ext_vector_type(8)))  float    v8f;

union AFrag { v16h v; _Float16 h[16]; };
union CFrag { v8f  v; float     f[8]; };

static __device__ __forceinline__ v8f wmma_f16(const v16h& a, const v16h& b, const v8f& c) {
    return __builtin_amdgcn_wmma_f32_16x16x32_f16(false, a, false, b, (short)0, c, false, false);
}

// LDS byte offset = low 32 bits of the generic pointer (flat aperture scheme,
// see ISA 10.2).
static __device__ __forceinline__ uint32_t lds_off(const void* p) {
    return (uint32_t)(uintptr_t)p;
}

// Async global -> LDS copy of 16 bytes per active lane (bypasses VGPRs;
// tracked with ASYNCcnt).
static __device__ __forceinline__ void async_cp_b128(uint32_t lds_addr, const float* gaddr) {
    asm volatile("global_load_async_to_lds_b128 %0, %1, off"
                 :: "v"(lds_addr), "v"(gaddr)
                 : "memory");
}

static __device__ __forceinline__ void wait_asynccnt0() {
    asm volatile("s_wait_asynccnt 0x0" ::: "memory");
}

// ---------------------------------------------------------------------------
// Generic GEMM:  Y[M,N] = act( X[M,K] * W[N,K]^T + bias[N] )
// Block = 128 threads (4 waves). 64x64 output tile, K stepped by 32.
// f32 tiles DMA'd to LDS asynchronously (double-buffered); f32->f16 conversion
// happens in the fragment build so it co-executes with the WMMA stream.
// Requires: K % 32 == 0, N % 64 == 0 (true at all call sites). M is guarded.
// ---------------------------------------------------------------------------
__global__ __launch_bounds__(128)
void wmma_gemm_bias(const float* __restrict__ X, const float* __restrict__ W,
                    const float* __restrict__ bias, float* __restrict__ Y,
                    int M, int N, int K, int relu)
{
    __shared__ alignas(16) float XsF[2][64 * 32];
    __shared__ alignas(16) float WsF[2][64 * 32];

    const int tid  = threadIdx.x;
    const int lane = tid & 31;
    const int wave = tid >> 5;
    const int row0 = blockIdx.x * 64;
    const int col0 = blockIdx.y * 64;
    const int wr   = (wave >> 1) * 32;   // wave row offset in tile
    const int wc   = (wave & 1) * 32;    // wave col offset in tile
    const int mrow = lane & 15;
    const int hi8  = (lane >> 4) * 8;    // A-frag K offset for upper half-wave
    const int kofB = (lane >> 4) * 16;   // B-frag K offset for upper half-wave

    // per-thread staging slot: 4 float4 chunks covering the 64x32 tile
    const int sr = tid >> 3;             // tile row 0..15 (+16 per chunk)
    const int sc = (tid & 7) * 4;        // tile col 0,4,..,28

    CFrag acc[2][2];
#pragma unroll
    for (int i = 0; i < 2; ++i)
#pragma unroll
        for (int j = 0; j < 2; ++j)
            acc[i][j].v = (v8f){0.f, 0.f, 0.f, 0.f, 0.f, 0.f, 0.f, 0.f};

    // ---- async stage of K-step k0 into buffer buf ----
    auto stage = [&](int buf, int k0) {
#pragma unroll
        for (int i = 0; i < 4; ++i) {
            const int r  = sr + i * 16;
            const int gr = row0 + r;
            if (gr < M)   // stale LDS for masked rows only feeds masked C rows
                async_cp_b128(lds_off(&XsF[buf][r * 32 + sc]),
                              X + (size_t)gr * K + k0 + sc);
            const int gw = col0 + r;
            if (gw < N)
                async_cp_b128(lds_off(&WsF[buf][r * 32 + sc]),
                              W + (size_t)gw * K + k0 + sc);
        }
    };

    stage(0, 0);
    wait_asynccnt0();
    __syncthreads();

    int cur = 0;
    for (int k0 = 0; k0 < K; k0 += 32) {
        if (k0 + 32 < K) stage(cur ^ 1, k0 + 32);   // overlap DMA with WMMA

        const float* xsb = &XsF[cur][0];
        const float* wsb = &WsF[cur][0];
        AFrag a[2], b[2];
#pragma unroll
        for (int t = 0; t < 2; ++t) {
            const int m = wr + t * 16 + mrow;
#pragma unroll
            for (int e = 0; e < 16; ++e) {
                const int k = e + (e & 8) + hi8;      // 16-bit A 16x32 layout
                a[t].h[e] = (_Float16)xsb[m * 32 + k];
            }
            const int n = wc + t * 16 + mrow;
#pragma unroll
            for (int e = 0; e < 16; ++e)              // B: same K order as A
                b[t].h[e] = (_Float16)wsb[n * 32 + e + (e & 8) + hi8];
        }
#pragma unroll
        for (int i = 0; i < 2; ++i)
#pragma unroll
            for (int j = 0; j < 2; ++j)
                acc[i][j].v = wmma_f16(a[i].v, b[j].v, acc[i][j].v);

        wait_asynccnt0();     // own async transfers landed in LDS
        __syncthreads();      // everyone's landed; readers of 'cur' done
        cur ^= 1;
    }

    // ---- epilogue: bias (+ReLU), guarded store ----
#pragma unroll
    for (int i = 0; i < 2; ++i)
#pragma unroll
        for (int j = 0; j < 2; ++j) {
            const int n  = col0 + wc + j * 16 + mrow;
            const float bv = bias[n];
#pragma unroll
            for (int r = 0; r < 8; ++r) {
                const int m = row0 + wr + i * 16 + r + hi8;
                if (m < M) {
                    float y = acc[i][j].f[r] + bv;
                    if (relu) y = fmaxf(y, 0.f);
                    Y[(size_t)m * N + n] = y;
                }
            }
        }
}

// ---------------------------------------------------------------------------
// Flash self-attention. One wave per (b, h, 16-query tile). DH==32 == WMMA K,
// so S-tile = one WMMA; P*V = two WMMAs (DH split 2x16 cols) with P zero-
// padded in K. Online softmax with shfl_xor row reductions.
// qk layout: [NTOK][512]  (cols 0..255 = Q heads, 256..511 = K heads)
// v  layout: [NTOK][256]
// ---------------------------------------------------------------------------
__global__ __launch_bounds__(32)
void flash_self_attn(const float* __restrict__ qk, const float* __restrict__ v,
                     float* __restrict__ o)
{
    __shared__ alignas(32) _Float16 P[16 * 16];

    const int lane = threadIdx.x;
    const int qt   = blockIdx.x;   // query tile 0..56
    const int h    = blockIdx.y;
    const int b    = blockIdx.z;
    const int mrow = lane & 15;
    const int hi8  = (lane >> 4) * 8;
    const int kofB = (lane >> 4) * 16;

    // Q fragment (pre-scaled)
    AFrag aQ;
    {
        int qm = qt * 16 + mrow; if (qm >= NQ) qm = NQ - 1;
        const float* qrow = qk + (size_t)(b * NQ + qm) * 512 + h * DH;
#pragma unroll
        for (int e = 0; e < 16; ++e) {
            const int k = e + (e & 8) + hi8;          // 0..31 == head dim
            aQ.h[e] = (_Float16)(qrow[k] * ATT_SCALE);
        }
    }

    float mi[8], li[8];
    CFrag oA, oB;
#pragma unroll
    for (int r = 0; r < 8; ++r) { mi[r] = -1e30f; li[r] = 0.f; oA.f[r] = 0.f; oB.f[r] = 0.f; }

    const int nKT = (NQ + 15) / 16;   // 57
    for (int kt = 0; kt < nKT; ++kt) {
        // K^T fragment: B[k][n] = K[kt*16+n][k]
        AFrag bK;
        const int kn  = kt * 16 + mrow;
        const bool knv = (kn < NQ);
        const int knc = knv ? kn : (NQ - 1);
        const float* krow = qk + (size_t)(b * NQ + knc) * 512 + 256 + h * DH;
#pragma unroll
        for (int e = 0; e < 16; ++e) bK.h[e] = (_Float16)krow[e + (e & 8) + hi8];

        CFrag S; S.v = (v8f){0.f, 0.f, 0.f, 0.f, 0.f, 0.f, 0.f, 0.f};
        S.v = wmma_f16(aQ.v, bK.v, S.v);
        if (!knv) {
#pragma unroll
            for (int r = 0; r < 8; ++r) S.f[r] = -1e30f;   // mask invalid keys
        }

        // online softmax (row = r + hi8, col = lane%16 spread over 16 lanes)
#pragma unroll
        for (int r = 0; r < 8; ++r) {
            float s = S.f[r];
            float rm = s;
#pragma unroll
            for (int off = 1; off < 16; off <<= 1) rm = fmaxf(rm, __shfl_xor(rm, off, 32));
            const float mnew = fmaxf(mi[r], rm);
            const float p    = __expf(s - mnew);
            float psum = p;
#pragma unroll
            for (int off = 1; off < 16; off <<= 1) psum += __shfl_xor(psum, off, 32);
            const float sc = __expf(mi[r] - mnew);
            li[r] = li[r] * sc + psum;
            mi[r] = mnew;
            oA.f[r] *= sc;
            oB.f[r] *= sc;
            P[(r + hi8) * 16 + mrow] = (_Float16)p;    // C-layout -> LDS
        }
        __syncthreads();

        // P as A-fragment (K padded 16->32 with zeros)
        AFrag aP;
#pragma unroll
        for (int e = 0; e < 16; ++e) {
            const int k = e + (e & 8) + hi8;
            aP.h[e] = (k < 16) ? P[mrow * 16 + k] : (_Float16)0.f;
        }
        // V fragments: two 16-column halves of DH
        AFrag bV0, bV1;
#pragma unroll
        for (int e = 0; e < 16; ++e) {
            const int k = e + (e & 8) + hi8;   // key index in tile (valid < 16)
            const int g = kt * 16 + k;
            float v0 = 0.f, v1 = 0.f;
            if (k < 16 && g < NQ) {
                const float* vrow = v + (size_t)(b * NQ + g) * 256 + h * DH;
                v0 = vrow[mrow];
                v1 = vrow[16 + mrow];
            }
            bV0.h[e] = (_Float16)v0;
            bV1.h[e] = (_Float16)v1;
        }
        oA.v = wmma_f16(aP.v, bV0.v, oA.v);
        oB.v = wmma_f16(aP.v, bV1.v, oB.v);
        __syncthreads();
    }

#pragma unroll
    for (int r = 0; r < 8; ++r) {
        const int m = qt * 16 + r + hi8;
        if (m < NQ) {
            const float inv = 1.f / li[r];
            float* orow = o + (size_t)(b * NQ + m) * 256 + h * DH;
            orow[mrow]      = oA.f[r] * inv;
            orow[16 + mrow] = oB.f[r] * inv;
        }
    }
}

// ---------------------------------------------------------------------------
// Multi-scale deformable attention sampling. One wave per (b,q,h); lane = DH
// channel, so value gathers are 128B-coalesced.
// ---------------------------------------------------------------------------
__device__ __forceinline__ float msda_gather(const float* __restrict__ vb,
                                             int st, int H, int W, int xi, int yi)
{
    if (xi < 0 || xi >= W || yi < 0 || yi >= H) return 0.f;
    return vb[(size_t)(st + yi * W + xi) * 256];
}

__global__ __launch_bounds__(256)
void msda_kernel(const float* __restrict__ value, const float* __restrict__ offs,
                 const float* __restrict__ attnl, const float* __restrict__ refp,
                 const float* __restrict__ vratio, float* __restrict__ oca)
{
    const int lane = threadIdx.x & 31;
    const int wid  = blockIdx.x * (blockDim.x >> 5) + (threadIdx.x >> 5);
    if (wid >= NTOK * HEADS) return;
    const int h = wid % HEADS;
    const int t = wid / HEADS;     // b*NQ + q
    const int b = t / NQ;

    // softmax over 16 (level,point) logits for this head
    float w[16];
    float mx = -1e30f;
#pragma unroll
    for (int i = 0; i < 16; ++i) {
        w[i] = attnl[(size_t)t * 128 + h * 16 + i];
        mx = fmaxf(mx, w[i]);
    }
    float s = 0.f;
#pragma unroll
    for (int i = 0; i < 16; ++i) { w[i] = __expf(w[i] - mx); s += w[i]; }
    const float inv = 1.f / s;

    const float rx = refp[(size_t)t * 3 + 0];
    const float rz = refp[(size_t)t * 3 + 2];

    const int Hs[4] = {92, 46, 23, 12};
    const int Wl_[4] = {160, 80, 40, 20};
    const int St[4] = {0, 14720, 18400, 19320};

    float acc = 0.f;
    const float* vb = value + (size_t)b * SEQLEN * 256 + h * DH + lane;

#pragma unroll
    for (int l = 0; l < LEVELS; ++l) {
        const int Hl = Hs[l], Wl = Wl_[l];
        const float rix = rx * vratio[(b * LEVELS + l) * 2 + 0];
        const float riy = rz * vratio[(b * LEVELS + l) * 2 + 1];
#pragma unroll
        for (int p = 0; p < POINTS; ++p) {
            const int oi = ((h * LEVELS + l) * POINTS + p) * 2;
            const float ox = offs[(size_t)t * 256 + oi + 0];
            const float oy = offs[(size_t)t * 256 + oi + 1];
            const float locx = rix + ox / (float)Wl;
            const float locy = riy + oy / (float)Hl;
            const float x = locx * (float)Wl - 0.5f;
            const float y = locy * (float)Hl - 0.5f;
            const float x0f = floorf(x), y0f = floorf(y);
            const float fx = x - x0f, fy = y - y0f;
            const int x0 = (int)x0f, y0 = (int)y0f;
            float samp = 0.f;
            samp += msda_gather(vb, St[l], Hl, Wl, x0,     y0    ) * (1.f - fx) * (1.f - fy);
            samp += msda_gather(vb, St[l], Hl, Wl, x0 + 1, y0    ) * fx * (1.f - fy);
            samp += msda_gather(vb, St[l], Hl, Wl, x0,     y0 + 1) * (1.f - fx) * fy;
            samp += msda_gather(vb, St[l], Hl, Wl, x0 + 1, y0 + 1) * fx * fy;
            acc += samp * (w[l * POINTS + p] * inv);
        }
    }
    oca[(size_t)t * 256 + h * DH + lane] = acc;
}

// ---------------------------------------------------------------------------
// out = LayerNorm(out + res) * w + b  — one block per token row (D=256)
// ---------------------------------------------------------------------------
__global__ __launch_bounds__(256)
void add_ln_kernel(float* __restrict__ out, const float* __restrict__ res,
                   const float* __restrict__ w, const float* __restrict__ b)
{
    __shared__ float red[256];
    const int t = blockIdx.x, i = threadIdx.x;
    const float x = out[(size_t)t * 256 + i] + res[(size_t)t * 256 + i];

    red[i] = x;
    __syncthreads();
#pragma unroll
    for (int s = 128; s > 0; s >>= 1) { if (i < s) red[i] += red[i + s]; __syncthreads(); }
    const float mu = red[0] * (1.f / 256.f);
    __syncthreads();

    const float d = x - mu;
    red[i] = d * d;
    __syncthreads();
#pragma unroll
    for (int s = 128; s > 0; s >>= 1) { if (i < s) red[i] += red[i + s]; __syncthreads(); }
    const float var = red[0] * (1.f / 256.f);

    out[(size_t)t * 256 + i] = d * rsqrtf(var + EPSV) * w[i] + b[i];
}

__global__ void add_kernel(const float* __restrict__ a, const float* __restrict__ b,
                           float* __restrict__ c, int n)
{
    const int i = blockIdx.x * blockDim.x + threadIdx.x;
    if (i < n) c[i] = a[i] + b[i];
}

__global__ void copy_kernel(float* __restrict__ dst, const float* __restrict__ src, int n)
{
    const int i = blockIdx.x * blockDim.x + threadIdx.x;
    if (i < n) dst[i] = src[i];
}

// ---------------------------------------------------------------------------
// Host orchestration (graph-capture safe: stream launches only)
// ---------------------------------------------------------------------------
extern "C" void kernel_launch(void* const* d_in, const int* in_sizes, int n_in,
                              void* d_out, int out_size, void* d_ws, size_t ws_size,
                              hipStream_t stream)
{
    const float* tgt      = (const float*)d_in[0];
    const float* refp     = (const float*)d_in[1];
    const float* src      = (const float*)d_in[2];
    const float* vratio   = (const float*)d_in[3];
    const float* qpos     = (const float*)d_in[4];
    const float* sa_in_w  = (const float*)d_in[5];
    const float* sa_in_b  = (const float*)d_in[6];
    const float* sa_out_w = (const float*)d_in[7];
    const float* sa_out_b = (const float*)d_in[8];
    const float* samp_w   = (const float*)d_in[9];
    const float* samp_b   = (const float*)d_in[10];
    const float* attn_w   = (const float*)d_in[11];
    const float* attn_b   = (const float*)d_in[12];
    const float* val_w    = (const float*)d_in[13];
    const float* val_b    = (const float*)d_in[14];
    const float* ca_out_w = (const float*)d_in[15];
    const float* ca_out_b = (const float*)d_in[16];
    const float* ffn1_w   = (const float*)d_in[17];
    const float* ffn1_b   = (const float*)d_in[18];
    const float* ffn2_w   = (const float*)d_in[19];
    const float* ffn2_b   = (const float*)d_in[20];
    const float* ln1_w    = (const float*)d_in[21];
    const float* ln1_b    = (const float*)d_in[22];
    const float* ln2_w    = (const float*)d_in[23];
    const float* ln2_b    = (const float*)d_in[24];
    const float* ln3_w    = (const float*)d_in[25];
    const float* ln3_b    = (const float*)d_in[26];

    float* ws = (float*)d_ws;
    size_t off = 0;
    float* out  = ws + off; off += (size_t)NTOK * 256;
    float* qbuf = ws + off; off += (size_t)NTOK * 256;
    float* qk   = ws + off; off += (size_t)NTOK * 512;
    float* vsa  = ws + off; off += (size_t)NTOK * 256;
    float* osa  = ws + off; off += (size_t)NTOK * 256;
    float* tmpA = ws + off; off += (size_t)NTOK * 1024;   // FFN hidden
    float* tmpB = ws + off; off += (size_t)NTOK * 256;    // projection outs
    float* val  = ws + off; off += (size_t)NSRC * 256;    // MSDA values
    float* offs = ws + off; off += (size_t)NTOK * 256;    // sampling offsets
    float* attw = ws + off; off += (size_t)NTOK * 128;    // MSDA attn logits
    float* oca  = ws + off; off += (size_t)NTOK * 256;

    const int NELEM = NTOK * 256;
    const dim3 blk128(128), blk256(256), blk32(32);
    const int gTok = (NTOK + 63) / 64;        // 57
    const int gSrc = (NSRC + 63) / 64;        // 1223

    copy_kernel<<<(NELEM + 255) / 256, blk256, 0, stream>>>(out, tgt, NELEM);

    for (int l = 0; l < NLAY; ++l) {
        const float* saw = sa_in_w  + (size_t)l * 768 * 256;
        const float* sab = sa_in_b  + (size_t)l * 768;
        const float* sow = sa_out_w + (size_t)l * 256 * 256;
        const float* sob = sa_out_b + (size_t)l * 256;
        const float* spw = samp_w   + (size_t)l * 256 * 256;
        const float* spb = samp_b   + (size_t)l * 256;
        const float* atw = attn_w   + (size_t)l * 128 * 256;
        const float* atb = attn_b   + (size_t)l * 128;
        const float* vw  = val_w    + (size_t)l * 256 * 256;
        const float* vb  = val_b    + (size_t)l * 256;
        const float* cow = ca_out_w + (size_t)l * 256 * 256;
        const float* cob = ca_out_b + (size_t)l * 256;
        const float* f1w = ffn1_w   + (size_t)l * 1024 * 256;
        const float* f1b = ffn1_b   + (size_t)l * 1024;
        const float* f2w = ffn2_w   + (size_t)l * 256 * 1024;
        const float* f2b = ffn2_b   + (size_t)l * 256;

        // ---- self-attention ----
        add_kernel<<<(NELEM + 255) / 256, blk256, 0, stream>>>(out, qpos, qbuf, NELEM);
        // Q,K projections (from out+qpos), V projection (from out)
        wmma_gemm_bias<<<dim3(gTok, 8), blk128, 0, stream>>>(qbuf, saw, sab, qk,
                                                             NTOK, 512, 256, 0);
        wmma_gemm_bias<<<dim3(gTok, 4), blk128, 0, stream>>>(out, saw + (size_t)512 * 256,
                                                             sab + 512, vsa,
                                                             NTOK, 256, 256, 0);
        flash_self_attn<<<dim3(57, HEADS, BSZ), blk32, 0, stream>>>(qk, vsa, osa);
        wmma_gemm_bias<<<dim3(gTok, 4), blk128, 0, stream>>>(osa, sow, sob, tmpB,
                                                             NTOK, 256, 256, 0);
        add_ln_kernel<<<NTOK, blk256, 0, stream>>>(out, tmpB,
                                                   ln1_w + (size_t)l * 256, ln1_b + (size_t)l * 256);

        // ---- deformable cross-attention ----
        add_kernel<<<(NELEM + 255) / 256, blk256, 0, stream>>>(out, qpos, qbuf, NELEM);
        wmma_gemm_bias<<<dim3(gSrc, 4), blk128, 0, stream>>>(src, vw, vb, val,
                                                             NSRC, 256, 256, 0);
        wmma_gemm_bias<<<dim3(gTok, 4), blk128, 0, stream>>>(qbuf, spw, spb, offs,
                                                             NTOK, 256, 256, 0);
        wmma_gemm_bias<<<dim3(gTok, 2), blk128, 0, stream>>>(qbuf, atw, atb, attw,
                                                             NTOK, 128, 256, 0);
        msda_kernel<<<NTOK, blk256, 0, stream>>>(val, offs, attw, refp, vratio, oca);
        wmma_gemm_bias<<<dim3(gTok, 4), blk128, 0, stream>>>(oca, cow, cob, tmpB,
                                                             NTOK, 256, 256, 0);
        add_ln_kernel<<<NTOK, blk256, 0, stream>>>(out, tmpB,
                                                   ln2_w + (size_t)l * 256, ln2_b + (size_t)l * 256);

        // ---- FFN ----
        wmma_gemm_bias<<<dim3(gTok, 16), blk128, 0, stream>>>(out, f1w, f1b, tmpA,
                                                              NTOK, 1024, 256, 1);
        wmma_gemm_bias<<<dim3(gTok, 4), blk128, 0, stream>>>(tmpA, f2w, f2b, tmpB,
                                                             NTOK, 256, 1024, 0);
        add_ln_kernel<<<NTOK, blk256, 0, stream>>>(out, tmpB,
                                                   ln3_w + (size_t)l * 256, ln3_b + (size_t)l * 256);
    }

    copy_kernel<<<(NELEM + 255) / 256, blk256, 0, stream>>>((float*)d_out, out, NELEM);
}
